// TinKerAttention_39204461477980
// MI455X (gfx1250) — matrix operand order repaired and checked by reading, not compiled, not run on hardware
//
#include <hip/hip_runtime.h>

// ---------------------------------------------------------------------------
// CDNA5 (gfx1250) wave32 WMMA attention block.
//   x[B,S,H] fp32 -> Q/K/V proj (bf16 WMMA, fp32 accum) -> RoPE -> flash
//   attention (bf16 WMMA, TDM + async-to-LDS tile staging) -> out proj -> fp32.
// Shapes fixed per reference: B=2 S=2048 H=2048 NH=16 NKV=4 HD=128.
// ---------------------------------------------------------------------------

typedef __attribute__((ext_vector_type(16))) __bf16    v16bf;
typedef __attribute__((ext_vector_type(8)))  float     v8f;
typedef __attribute__((ext_vector_type(4)))  unsigned  v4u;
typedef __attribute__((ext_vector_type(8)))  unsigned  v8u;

union V16BF { v16bf v; uint4 q[2]; };

__device__ __forceinline__ v8f vzero8() {
  v8f z;
#pragma unroll
  for (int i = 0; i < 8; ++i) z[i] = 0.0f;
  return z;
}

__device__ __forceinline__ v8f wmma_bf16(v16bf a, v16bf b, v8f c) {
  // D = A(16x32 bf16) * B(32x16 bf16) + C(16x16 f32)
  return __builtin_amdgcn_wmma_f32_16x16x32_bf16(false, a, false, b,
                                                 (short)0, c, false, false);
}

__device__ __forceinline__ void cvt16(const float4 u0, const float4 u1,
                                      const float4 u2, const float4 u3,
                                      v16bf& d) {
  d[0]=(__bf16)u0.x; d[1]=(__bf16)u0.y; d[2]=(__bf16)u0.z; d[3]=(__bf16)u0.w;
  d[4]=(__bf16)u1.x; d[5]=(__bf16)u1.y; d[6]=(__bf16)u1.z; d[7]=(__bf16)u1.w;
  d[8]=(__bf16)u2.x; d[9]=(__bf16)u2.y; d[10]=(__bf16)u2.z; d[11]=(__bf16)u2.w;
  d[12]=(__bf16)u3.x; d[13]=(__bf16)u3.y; d[14]=(__bf16)u3.z; d[15]=(__bf16)u3.w;
}

// Async copy of one 16-byte chunk global -> LDS (GLOBAL_LOAD_ASYNC_TO_LDS_B128,
// GV mode). Tracked by ASYNCcnt.
__device__ __forceinline__ void async_copy_b128(unsigned lds_off,
                                                const void* gptr) {
  asm volatile("global_load_async_to_lds_b128 %0, %1, off"
               :
               : "v"(lds_off), "v"((unsigned long long)(uintptr_t)gptr)
               : "memory");
}

__device__ __forceinline__ void wait_asynccnt0() {
  asm volatile("s_wait_asynccnt 0x0" ::: "memory");
}

// Generic shared-memory pointer -> 32-bit LDS byte offset (aperture rule:
// the LDS offset is the low 32 address bits).
__device__ __forceinline__ unsigned lds_off32(const void* p) {
  return (unsigned)(uintptr_t)p;
}

// ---------------------------------------------------------------------------
// Tensor Data Mover: 2D tile load global -> LDS with row padding.
// D# built per CDNA5 ISA 8.3/8.4:
//   group0: [1:0]=count=1, [63:32]=lds_addr, [120:64]=global_addr,
//           [127:126]=type=2
//   group1: [17:16]=data_size(1 -> 2B), [20]=pad_enable, [24:22]=pad_interval,
//           [31:25]=pad_amount, [79:48]=tensor_dim0, [111:80]=tensor_dim1,
//           [127:112]=tile_dim0, [143:128]=tile_dim1, [207:160]=dim0_stride
//   groups2/3: zero (tile_dim3/4 unused)
// pad_interval code c -> 2^(c+1) DWORDs between pads; pad_amount code a ->
// (a+1) DWORDs inserted.  Issued once per workgroup (wave 0); completion via
// TENSORcnt (s_wait_tensorcnt) before the workgroup barrier.
// ---------------------------------------------------------------------------
__device__ __forceinline__ void tdm_load_2d(unsigned lds_addr, const void* gaddr,
                                            unsigned tensor_d0, unsigned tensor_d1,
                                            unsigned tile_d0, unsigned tile_d1,
                                            unsigned stride_d0,
                                            unsigned pad_interval_code,
                                            unsigned pad_amount_code) {
  const unsigned long long ga = (unsigned long long)(uintptr_t)gaddr;
  v4u g0;
  g0[0] = 1u;                                         // count = 1 (valid D#)
  g0[1] = lds_addr;                                   // LDS byte address
  g0[2] = (unsigned)(ga & 0xffffffffu);               // global_addr[31:0]
  g0[3] = (unsigned)((ga >> 32) & 0x01ffffffu)        // global_addr[56:32]
          | (2u << 30);                               // type = 2 ("image")
  v8u g1;
  g1[0] = (1u << 16)                                  // data_size = 2 bytes
          | (1u << 20)                                // pad_enable
          | (pad_interval_code << 22)
          | (pad_amount_code << 25);
  g1[1] = (tensor_d0 & 0xffffu) << 16;                // tensor_dim0[15:0]
  g1[2] = (tensor_d0 >> 16) | ((tensor_d1 & 0xffffu) << 16);
  g1[3] = (tensor_d1 >> 16) | (tile_d0 << 16);        // tile_dim0
  g1[4] = tile_d1 & 0xffffu;                          // tile_dim1 (dim2 = 0)
  g1[5] = stride_d0;                                  // dim0 stride (elements)
  g1[6] = 0u;
  g1[7] = 0u;
  const v4u gz = {0u, 0u, 0u, 0u};
  asm volatile("tensor_load_to_lds %0, %1, %2, %3"
               :
               : "s"(g0), "s"(g1), "s"(gz), "s"(gz)
               : "memory");
}

// ---------------------------------------------------------------------------
// Y[m,n] = sum_k A[m,k] * W[n,k]   (K = 2048 fixed, "NT" gemm)
// Workgroup = 256 threads = 8 waves (4 in M x 2 in N), tile 128M x 128N.
// Each wave: 32M x 64N = 2x4 accumulators of 16x16, K-step 32.
// MODE 0: bf16 out, head-major  [b][h][s][128]   (Q, K)
// MODE 1: bf16 out, transposed  [b][h][128][s]   (V^T)
// MODE 2: fp32 out, token-major [m][N]           (final output)
// ---------------------------------------------------------------------------
template <typename TA, int MODE>
__global__ __launch_bounds__(256) void gemm_xwT(const TA* __restrict__ A,
                                                const float* __restrict__ W,
                                                void* __restrict__ Out,
                                                int N) {
  const int  K    = 2048;
  const int  tid  = threadIdx.x;
  const int  lane = tid & 31;
  const int  w    = tid >> 5;
  const int  wm   = w & 3, wn = w >> 2;
  const int  l15  = lane & 15, hi = lane >> 4;
  const long mWave = (long)blockIdx.y * 128 + wm * 32;
  const long nWave = (long)blockIdx.x * 128 + wn * 64;

  v8f acc[2][4];
#pragma unroll
  for (int i = 0; i < 2; ++i)
#pragma unroll
    for (int j = 0; j < 4; ++j) acc[i][j] = vzero8();

  for (int k0 = 0; k0 < K; k0 += 32) {
    // L2 prefetch of next K-step fragments (global_prefetch_b8).
    if (k0 + 32 < K) {
      __builtin_prefetch(A + (mWave + l15) * (long)K + k0 + 32 + hi * 8, 0, 1);
      __builtin_prefetch(W + (nWave + l15) * (long)K + k0 + 32 + hi * 16, 0, 1);
    }
    // A fragments: 16x32, lane l<16 holds row l, K {0..7,16..23} (rel),
    // lane l>=16 holds row l-16, K {8..15,24..31}.
    V16BF a[2];
#pragma unroll
    for (int i = 0; i < 2; ++i) {
      if constexpr (__is_same(TA, float)) {
        const float4* p4 =
            (const float4*)(A + (mWave + i * 16 + l15) * (long)K + k0 + hi * 8);
        cvt16(p4[0], p4[1], p4[4], p4[5], a[i].v);
      } else {
        const __bf16* p = A + (mWave + i * 16 + l15) * (long)K + k0 + hi * 8;
        a[i].q[0] = *(const uint4*)p;
        a[i].q[1] = *(const uint4*)(p + 16);
      }
    }
    // B fragments: 32x16, lane = column n, K {0..7,16..23} (hi half: +8).
    V16BF b[4];
#pragma unroll
    for (int j = 0; j < 4; ++j) {
      const float4* p4 =
          (const float4*)(W + (nWave + j * 16 + l15) * (long)K + k0 + hi * 8);
      cvt16(p4[0], p4[1], p4[4], p4[5], b[j].v);
    }
#pragma unroll
    for (int i = 0; i < 2; ++i)
#pragma unroll
      for (int j = 0; j < 4; ++j)
        acc[i][j] = wmma_bf16(a[i].v, b[j].v, acc[i][j]);
  }

  // C/D layout: VGPR r, lanes 0-15 -> row r, lanes 16-31 -> row r+8.
#pragma unroll
  for (int i = 0; i < 2; ++i)
#pragma unroll
    for (int j = 0; j < 4; ++j)
#pragma unroll
      for (int r = 0; r < 8; ++r) {
        const long m   = mWave + i * 16 + hi * 8 + r;
        const long n   = nWave + j * 16 + l15;
        const float val = acc[i][j][r];
        if constexpr (MODE == 2) {
          ((float*)Out)[m * (long)N + n] = val;
        } else {
          const long bb = m >> 11, s = m & 2047;   // S = 2048
          const long h = n >> 7, d = n & 127;      // HD = 128
          const long H = (long)N >> 7;
          if constexpr (MODE == 0)
            ((__bf16*)Out)[((bb * H + h) * 2048 + s) * 128 + d] = (__bf16)val;
          else
            ((__bf16*)Out)[((bb * H + h) * 128 + d) * 2048 + s] = (__bf16)val;
        }
      }
}

// ---------------------------------------------------------------------------
// RoPE in place on bf16 head-major tensor [bh][s][128], rotate-half pairs
// (d, d+64). total = BH * S * 64 threads of work.
// ---------------------------------------------------------------------------
__global__ __launch_bounds__(256) void rope_inplace(__bf16* __restrict__ t,
                                                    const float* __restrict__ cosT,
                                                    const float* __restrict__ sinT,
                                                    int total) {
  const int i = blockIdx.x * 256 + threadIdx.x;
  if (i >= total) return;
  const int  d    = i & 63;
  const int  s    = (i >> 6) & 2047;
  const long base = ((long)(i >> 17) * 2048 + s) * 128;
  const float c  = cosT[s * 128 + d];
  const float sn = sinT[s * 128 + d];
  const float v0 = (float)t[base + d];
  const float v1 = (float)t[base + d + 64];
  t[base + d]      = (__bf16)(v0 * c - v1 * sn);
  t[base + d + 64] = (__bf16)(v1 * c + v0 * sn);
}

// ---------------------------------------------------------------------------
// Flash attention, causal, GQA (group=4). Grid: (S/128, B*NH).
// Block 256 = 8 waves; wave w owns q rows [q0+16w, q0+16w+16).
// K tile staged by the Tensor Data Mover (wave 0, padded rows via D# pad),
// V^T tile staged by GLOBAL_LOAD_ASYNC_TO_LDS_B128; both completed before
// the workgroup barrier (TENSORcnt / ASYNCcnt).
// ---------------------------------------------------------------------------
__global__ __launch_bounds__(256) void flash_fwd(const __bf16* __restrict__ Q,
                                                 const __bf16* __restrict__ Kg,
                                                 const __bf16* __restrict__ VT,
                                                 __bf16* __restrict__ AttnOut) {
  const int S = 2048, HD = 128, NH = 16, NKV = 4;
  const int qt  = blockIdx.x;
  const int bh  = blockIdx.y;
  const int b   = bh / NH, h = bh % NH, kvh = h / (NH / NKV);
  const int q0  = qt * 128;
  const int tid = threadIdx.x, lane = tid & 31, w = tid >> 5;
  const int l15 = lane & 15, hi = lane >> 4;

  __shared__ __bf16 Ks[64][136];    // keys x feat (row = 64 DW + 4 DW pad)
  __shared__ __bf16 Vs[128][72];    // feat x keys (row = 32 DW + 4 DW pad)
  __shared__ __bf16 Ps[8][16][72];  // per-wave probability staging

  const __bf16* Qp = Q  + ((long)(b * NH + h) * S + q0) * HD;
  const __bf16* Kp = Kg + (long)(b * NKV + kvh) * S * HD;
  const __bf16* Vp = VT + (long)(b * NKV + kvh) * HD * S;

  // Q A-fragments for this wave's 16 rows (K dim = 128 feats, 4 steps of 32).
  V16BF qa[4];
#pragma unroll
  for (int c = 0; c < 4; ++c) {
    const __bf16* p = Qp + (long)(w * 16 + l15) * HD + c * 32 + hi * 8;
    qa[c].q[0] = *(const uint4*)p;
    qa[c].q[1] = *(const uint4*)(p + 16);
  }

  float mrow[8], lrow[8];
  v8f o[8];
#pragma unroll
  for (int r = 0; r < 8; ++r) { mrow[r] = -3.0e38f; lrow[r] = 0.0f; }
#pragma unroll
  for (int j = 0; j < 8; ++j) o[j] = vzero8();

  const float scale = 0.08838834764831845f;  // 1/sqrt(128)
  const int nkt = q0 / 64 + 2;               // keys up to q0+127
  for (int kt = 0; kt < nkt; ++kt) {
    const int k0 = kt * 64;
    __syncthreads();  // previous tile fully consumed by all waves

    // K tile 64x128 via TDM (one DMA, rows padded 64+4 DWORDs -> stride 136).
    if (w == 0) {
      tdm_load_2d(lds_off32(&Ks[0][0]), Kp + (long)k0 * HD,
                  /*tensor_d0=*/128, /*tensor_d1=*/64,
                  /*tile_d0=*/128, /*tile_d1=*/64,
                  /*stride_d0=*/128,
                  /*pad_interval: 2^(5+1)=64 DW*/ 5u,
                  /*pad_amount: 3+1=4 DW*/ 3u);
    }
    // V^T tile 128x64 via async-to-LDS b128 (1024 chunks, 4 per thread).
#pragma unroll
    for (int it = 0; it < 4; ++it) {
      const int idx = tid + it * 256;
      const int d = idx >> 3, c = (idx & 7) << 3;      // 8 chunks per V^T row
      async_copy_b128(lds_off32(&Vs[d][c]),
                      &Vp[(long)d * S + k0 + c]);
    }
    wait_asynccnt0();                          // this wave's async copies done
    if (w == 0) __builtin_amdgcn_s_wait_tensorcnt(0);  // TDM done
    __syncthreads();                           // all staging visible

    // scores = Q * K^T : B[k=feat][n=key] = Ks[key][feat]
    v8f sc[4];
#pragma unroll
    for (int j = 0; j < 4; ++j) {
      v8f s_ = vzero8();
#pragma unroll
      for (int c = 0; c < 4; ++c) {
        V16BF kb;
        const __bf16* p = &Ks[j * 16 + l15][c * 32 + hi * 8];
        kb.q[0] = *(const uint4*)p;
        kb.q[1] = *(const uint4*)(p + 16);
        s_ = wmma_bf16(qa[c].v, kb.v, s_);
      }
      sc[j] = s_;
    }

    // Scale + causal mask + online softmax.
    const bool needMask = (k0 + 63) > (q0 + w * 16);
    float rmax[8];
#pragma unroll
    for (int r = 0; r < 8; ++r) rmax[r] = -3.0e38f;
#pragma unroll
    for (int j = 0; j < 4; ++j)
#pragma unroll
      for (int r = 0; r < 8; ++r) {
        float v = sc[j][r] * scale;
        if (needMask) {
          const int col = k0 + j * 16 + l15;
          const int row = q0 + w * 16 + hi * 8 + r;
          if (col > row) v = -3.0e38f;
        }
        sc[j][r] = v;
        rmax[r]  = fmaxf(rmax[r], v);
      }
    // Row reductions across 16 lanes within each half-wave (wave32).
#pragma unroll
    for (int off = 1; off < 16; off <<= 1)
#pragma unroll
      for (int r = 0; r < 8; ++r)
        rmax[r] = fmaxf(rmax[r], __shfl_xor(rmax[r], off, 32));

    float alpha[8], rsum[8];
#pragma unroll
    for (int r = 0; r < 8; ++r) {
      const float mn = fmaxf(mrow[r], rmax[r]);
      alpha[r] = __expf(mrow[r] - mn);
      mrow[r]  = mn;
      rsum[r]  = 0.0f;
    }
#pragma unroll
    for (int j = 0; j < 4; ++j)
#pragma unroll
      for (int r = 0; r < 8; ++r) {
        const float p = __expf(sc[j][r] - mrow[r]);
        sc[j][r] = p;
        rsum[r] += p;
      }
#pragma unroll
    for (int off = 1; off < 16; off <<= 1)
#pragma unroll
      for (int r = 0; r < 8; ++r) rsum[r] += __shfl_xor(rsum[r], off, 32);
#pragma unroll
    for (int r = 0; r < 8; ++r) lrow[r] = lrow[r] * alpha[r] + rsum[r];
#pragma unroll
    for (int j = 0; j < 8; ++j)
#pragma unroll
      for (int r = 0; r < 8; ++r) o[j][r] *= alpha[r];

    // P: C-layout -> per-wave LDS -> A-layout (same-wave DS ops are in order).
#pragma unroll
    for (int j = 0; j < 4; ++j)
#pragma unroll
      for (int r = 0; r < 8; ++r)
        Ps[w][hi * 8 + r][j * 16 + l15] = (__bf16)sc[j][r];

    // O += P(16x64) * V(64x128): B[k=key][n=feat] = Vs[feat][key].
#pragma unroll
    for (int kk = 0; kk < 64; kk += 32) {
      V16BF pa;
      {
        const __bf16* p = &Ps[w][l15][kk + hi * 8];
        pa.q[0] = *(const uint4*)p;
        pa.q[1] = *(const uint4*)(p + 16);
      }
#pragma unroll
      for (int j = 0; j < 8; ++j) {
        V16BF vb;
        const __bf16* p = &Vs[j * 16 + l15][kk + hi * 8];
        vb.q[0] = *(const uint4*)p;
        vb.q[1] = *(const uint4*)(p + 16);
        o[j] = wmma_bf16(pa.v, vb.v, o[j]);
      }
    }
  }

  // Normalize and store token-major bf16 [b*S+s][2048] for the Wo gemm.
#pragma unroll
  for (int r = 0; r < 8; ++r) lrow[r] = 1.0f / lrow[r];
#pragma unroll
  for (int j = 0; j < 8; ++j)
#pragma unroll
    for (int r = 0; r < 8; ++r) {
      const long m = q0 + w * 16 + hi * 8 + r;
      const long n = (long)h * 128 + j * 16 + l15;
      AttnOut[((long)b * S + m) * 2048 + n] = (__bf16)(o[j][r] * lrow[r]);
    }
}

// ---------------------------------------------------------------------------
// Host launcher. Workspace layout (bf16): Q[2*16*2048*128] | K[2*4*2048*128]
// | V^T[2*4*128*2048] | attn[4096*2048]  -> 40 MB total.
// ---------------------------------------------------------------------------
extern "C" void kernel_launch(void* const* d_in, const int* in_sizes, int n_in,
                              void* d_out, int out_size, void* d_ws, size_t ws_size,
                              hipStream_t stream) {
  (void)in_sizes; (void)n_in; (void)out_size; (void)ws_size;
  const float* x    = (const float*)d_in[0];
  const float* Wq   = (const float*)d_in[1];
  const float* Wk   = (const float*)d_in[2];
  const float* Wv   = (const float*)d_in[3];
  const float* Wo   = (const float*)d_in[4];
  const float* cosT = (const float*)d_in[5];
  const float* sinT = (const float*)d_in[6];

  const long B = 2, S = 2048, NH = 16, NKV = 4, HD = 128;

  __bf16* qb = (__bf16*)d_ws;
  __bf16* kb = qb + B * NH * S * HD;
  __bf16* vt = kb + B * NKV * S * HD;
  __bf16* at = vt + B * NKV * S * HD;

  const dim3 blk(256);
  // QKV projections (M=4096, K=2048). Tiles: grid.x = N/128, grid.y = M/128.
  gemm_xwT<float, 0><<<dim3(16, 32), blk, 0, stream>>>(x, Wq, (void*)qb, 2048);
  gemm_xwT<float, 0><<<dim3(4, 32),  blk, 0, stream>>>(x, Wk, (void*)kb, 512);
  gemm_xwT<float, 1><<<dim3(4, 32),  blk, 0, stream>>>(x, Wv, (void*)vt, 512);
  // RoPE on Q and K.
  rope_inplace<<<dim3((unsigned)(B * NH * S * 64 / 256)), blk, 0, stream>>>(
      qb, cosT, sinT, (int)(B * NH * S * 64));
  rope_inplace<<<dim3((unsigned)(B * NKV * S * 64 / 256)), blk, 0, stream>>>(
      kb, cosT, sinT, (int)(B * NKV * S * 64));
  // Flash attention: grid (S/128, B*NH).
  flash_fwd<<<dim3(16, 32), blk, 0, stream>>>(qb, kb, vt, at);
  // Output projection -> fp32 d_out.
  gemm_xwT<__bf16, 2><<<dim3(16, 32), blk, 0, stream>>>(at, Wo, d_out, 2048);
}
